// ExpertLinear_50002009260704
// MI455X (gfx1250) — matrix operand order repaired and checked by reading, not compiled, not run on hardware
//
#include <hip/hip_runtime.h>

// ---------------- problem constants (match reference) ----------------
#define N_TOK   8192
#define TOP_K   2
#define N_EXP   8
#define D_IN    1024
#define D_OUT   1024
#define NG      (N_TOK * TOP_K)          // 16384 grouped rows

#define BM      128
#define BN      128
#define BK      32
#define LDK     (BK + 8)                 // padded LDS k-stride
#define KSTEPS  (D_IN / BK)
#define MAX_ROW_TILES (NG / BM + N_EXP)

typedef unsigned short u16;
typedef __attribute__((ext_vector_type(16))) __bf16 v16bf;
typedef __attribute__((ext_vector_type(8)))  float  v8f;
struct Frag32 { uint4 a, b; };           // 32B == v16bf

// ---------------- bf16 split helpers (RNE) ----------------
__device__ inline u16 f32_to_bf16_rne(float f) {
    unsigned u = __builtin_bit_cast(unsigned, f);
    unsigned r = u + 0x7FFFu + ((u >> 16) & 1u);
    return (u16)(r >> 16);
}
__device__ inline float bf16_to_f32(u16 h) {
    return __builtin_bit_cast(float, ((unsigned)h) << 16);
}
__device__ inline void split_bf16(float v, u16& hi, u16& lo) {
    hi = f32_to_bf16_rne(v);
    lo = f32_to_bf16_rne(v - bf16_to_f32(hi));
}
__device__ inline unsigned pack2(u16 a, u16 b) { return (unsigned)a | ((unsigned)b << 16); }

// generic LDS pointer -> 32-bit LDS byte offset (flat LDS aperture: addr[31:0])
__device__ inline unsigned lds_off(const void* p) { return (unsigned)(size_t)p; }

// async 16B copy global -> LDS (GVS mode: saddr base + 32-bit per-lane offset)
__device__ inline void async_b128(unsigned lds_byte, const u16* sbase, unsigned voff_bytes) {
    asm volatile("global_load_async_to_lds_b128 %0, %1, %2"
                 :: "v"(lds_byte), "v"(voff_bytes), "s"(sbase)
                 : "memory");
}
__device__ inline void wait_async0() { asm volatile("s_wait_asynccnt 0x0" ::: "memory"); }

__device__ inline v16bf ld_frag(const u16* base, int k0, int k1) {
    Frag32 f;
    f.a = *(const uint4*)(base + k0);
    f.b = *(const uint4*)(base + k1);
    return __builtin_bit_cast(v16bf, f);
}

// ---------------- prep 1: split x -> bf16 hi/lo planes (+ zero row) ----------------
__global__ __launch_bounds__(256)
void moe_convert_x(const float* __restrict__ x, u16* __restrict__ xhi, u16* __restrict__ xlo) {
    int i = blockIdx.x * 256 + threadIdx.x;           // one float4 / thread; covers N_TOK+1 rows
    uint2 ph = make_uint2(0u, 0u), pl = make_uint2(0u, 0u);
    if (i < N_TOK * (D_IN / 4)) {
        float4 v = ((const float4*)x)[i];
        u16 h0,h1,h2,h3,l0,l1,l2,l3;
        split_bf16(v.x,h0,l0); split_bf16(v.y,h1,l1);
        split_bf16(v.z,h2,l2); split_bf16(v.w,h3,l3);
        ph.x = pack2(h0,h1); ph.y = pack2(h2,h3);
        pl.x = pack2(l0,l1); pl.y = pack2(l2,l3);
    }
    ((uint2*)xhi)[i] = ph;                            // row N_TOK stays zero (masked rows)
    ((uint2*)xlo)[i] = pl;
}

// ---------------- prep 2: split + transpose weights -> [e][n][k] bf16 planes ----------------
__global__ __launch_bounds__(256)
void moe_convert_w(const float* __restrict__ w, u16* __restrict__ whiT, u16* __restrict__ wloT) {
    __shared__ alignas(16) u16 thi[64][72];
    __shared__ alignas(16) u16 tlo[64][72];
    const int e = blockIdx.z, ktile = blockIdx.x, ntile = blockIdx.y;
    const int tid = threadIdx.x;
    const float* we = w + (size_t)e * D_IN * D_OUT;
    #pragma unroll
    for (int t = 0; t < 4; ++t) {                     // load 64k x 64n coalesced
        int idx = tid + 256 * t;
        int kk = idx >> 4, n4 = idx & 15;
        float4 v = *(const float4*)&we[(size_t)(ktile*64 + kk) * D_OUT + ntile*64 + n4*4];
        float vv[4] = { v.x, v.y, v.z, v.w };
        #pragma unroll
        for (int j = 0; j < 4; ++j) {
            u16 hi, lo; split_bf16(vv[j], hi, lo);
            thi[n4*4 + j][kk] = hi;
            tlo[n4*4 + j][kk] = lo;
        }
    }
    __syncthreads();
    u16* oh = whiT + (size_t)e * D_OUT * D_IN;
    u16* ol = wloT + (size_t)e * D_OUT * D_IN;
    #pragma unroll
    for (int t = 0; t < 2; ++t) {                     // write n-major, k-contiguous, coalesced
        int idx = tid + 256 * t;
        int n = idx >> 3, c = idx & 7;
        size_t o = (size_t)(ntile*64 + n) * D_IN + ktile*64 + c*8;
        *(uint4*)&oh[o] = *(const uint4*)&thi[n][c*8];
        *(uint4*)&ol[o] = *(const uint4*)&tlo[n][c*8];
    }
}

// ---------------- prep 3: invert scatter permutation ----------------
__global__ void moe_build_inv(const int* __restrict__ ssi, int* __restrict__ inv, int n) {
    int i = blockIdx.x * blockDim.x + threadIdx.x;
    if (i < n) inv[ssi[i]] = i;
}

// ---------------- grouped GEMM: async double-buffered bf16-split WMMA ----------------
__global__ __launch_bounds__(256)
void moe_grouped_gemm(const u16* __restrict__ xhi, const u16* __restrict__ xlo,
                      const u16* __restrict__ whiT, const u16* __restrict__ wloT,
                      const int* __restrict__ ssi, const int* __restrict__ eoff,
                      const float* __restrict__ gates, float* __restrict__ y)
{
    __shared__ alignas(16) u16 sAhi[2][BM][LDK];
    __shared__ alignas(16) u16 sAlo[2][BM][LDK];
    __shared__ alignas(16) u16 sBhi[2][BN][LDK];   // pre-transposed: [n][k]
    __shared__ alignas(16) u16 sBlo[2][BN][LDK];
    __shared__ float sGate[BM];

    // map blockIdx.x -> (expert, rowStart, segEnd) over sorted segments
    int tile = blockIdx.x;
    int prev = 0, expert = -1, rowStart = 0, segEnd = 0;
    #pragma unroll
    for (int e = 0; e < N_EXP; ++e) {
        int end = eoff[e];
        int len = end - prev;
        int nt  = (len + BM - 1) / BM;
        if (expert < 0) {
            if (tile < nt) { expert = e; rowStart = prev + tile * BM; segEnd = end; }
            else           { tile -= nt; }
        }
        prev = end;
    }
    if (expert < 0) return;

    const u16* wh = whiT + (size_t)expert * D_IN * D_OUT;
    const u16* wl = wloT + (size_t)expert * D_IN * D_OUT;
    const int colBase = blockIdx.y * BN;

    const int tid = threadIdx.x, lane = tid & 31, wave = tid >> 5;
    const int half = lane >> 4, l16 = lane & 15;
    const int waveRow = (wave >> 1) * 32;   // 4 row bands of 32
    const int waveCol = (wave & 1)  * 64;   // 2 col bands of 64

    // per-thread staging coordinates: rows r0 and r0+64, 16B chunk cch
    const int r0  = tid >> 2;               // 0..63
    const int cch = tid & 3;                // 0..3
    int tok0, tok1;                         // N_TOK == zero row for masked rows
    { int gr = rowStart + r0;      tok0 = (gr < segEnd) ? ssi[gr] / TOP_K : N_TOK; }
    { int gr = rowStart + 64 + r0; tok1 = (gr < segEnd) ? ssi[gr] / TOP_K : N_TOK; }

    if (tid < BM) {
        int gr = rowStart + tid;
        float gg = 0.0f;
        if (gr < segEnd) gg = gates[ssi[gr]];
        sGate[tid] = gg;
    }

    auto stage = [&](int kt, int buf) {
        const unsigned kb = (unsigned)(kt * BK + cch * 8) * 2u;   // k byte offset
        const unsigned co = (unsigned)cch * 16u;                  // LDS chunk offset
        const unsigned ax0 = (unsigned)tok0 * (D_IN * 2u) + kb;
        const unsigned ax1 = (unsigned)tok1 * (D_IN * 2u) + kb;
        async_b128(lds_off(&sAhi[buf][r0     ][0]) + co, xhi, ax0);
        async_b128(lds_off(&sAhi[buf][r0 + 64][0]) + co, xhi, ax1);
        async_b128(lds_off(&sAlo[buf][r0     ][0]) + co, xlo, ax0);
        async_b128(lds_off(&sAlo[buf][r0 + 64][0]) + co, xlo, ax1);
        const unsigned bn0 = (unsigned)(colBase + r0     ) * (D_IN * 2u) + kb;
        const unsigned bn1 = (unsigned)(colBase + r0 + 64) * (D_IN * 2u) + kb;
        async_b128(lds_off(&sBhi[buf][r0     ][0]) + co, wh, bn0);
        async_b128(lds_off(&sBhi[buf][r0 + 64][0]) + co, wh, bn1);
        async_b128(lds_off(&sBlo[buf][r0     ][0]) + co, wl, bn0);
        async_b128(lds_off(&sBlo[buf][r0 + 64][0]) + co, wl, bn1);
    };

    v8f acc[2][4];
    #pragma unroll
    for (int rt = 0; rt < 2; ++rt)
        #pragma unroll
        for (int ct = 0; ct < 4; ++ct)
            acc[rt][ct] = (v8f)(0.0f);

    stage(0, 0);
    for (int kt = 0; kt < KSTEPS; ++kt) {
        const int cur = kt & 1;
        wait_async0();          // my async writes into buf[cur] done
        __syncthreads();        // everyone's writes visible; buf[cur^1] reads retired
        if (kt + 1 < KSTEPS) stage(kt + 1, cur ^ 1);

        // fragments per ISA 16-bit WMMA layouts
        v16bf ahi[2], alo[2], bhi[4], blo[4];
        #pragma unroll
        for (int rt = 0; rt < 2; ++rt) {
            const u16* ph = &sAhi[cur][waveRow + rt * 16 + l16][0];
            const u16* pl = &sAlo[cur][waveRow + rt * 16 + l16][0];
            ahi[rt] = ld_frag(ph, half * 8, 16 + half * 8);
            alo[rt] = ld_frag(pl, half * 8, 16 + half * 8);
        }
        #pragma unroll
        for (int ct = 0; ct < 4; ++ct) {
            const u16* ph = &sBhi[cur][waveCol + ct * 16 + l16][0];
            const u16* pl = &sBlo[cur][waveCol + ct * 16 + l16][0];
            bhi[ct] = ld_frag(ph, half * 8, 16 + half * 8);
            blo[ct] = ld_frag(pl, half * 8, 16 + half * 8);
        }

        // 3-term bf16 split: hi*hi + hi*lo + lo*hi, f32 accumulate
        #pragma unroll
        for (int rt = 0; rt < 2; ++rt)
            #pragma unroll
            for (int ct = 0; ct < 4; ++ct) {
                acc[rt][ct] = __builtin_amdgcn_wmma_f32_16x16x32_bf16(
                    false, ahi[rt], false, bhi[ct], (short)0, acc[rt][ct], false, false);
                acc[rt][ct] = __builtin_amdgcn_wmma_f32_16x16x32_bf16(
                    false, ahi[rt], false, blo[ct], (short)0, acc[rt][ct], false, false);
                acc[rt][ct] = __builtin_amdgcn_wmma_f32_16x16x32_bf16(
                    false, alo[rt], false, bhi[ct], (short)0, acc[rt][ct], false, false);
            }
    }

    // epilogue: gate-scale, store grouped y (each valid row written exactly once)
    #pragma unroll
    for (int rt = 0; rt < 2; ++rt)
        #pragma unroll
        for (int ct = 0; ct < 4; ++ct)
            #pragma unroll
            for (int r = 0; r < 8; ++r) {
                int lrow = waveRow + rt * 16 + r + 8 * half;   // C layout: M = r + 8*half
                int gr = rowStart + lrow;
                if (gr < segEnd) {
                    int col = colBase + waveCol + ct * 16 + l16;
                    y[(size_t)gr * D_OUT + col] = sGate[lrow] * acc[rt][ct][r];
                }
            }
}

// ---------------- deterministic gated combine ----------------
__global__ __launch_bounds__(256)
void moe_combine(const float* __restrict__ y, const int* __restrict__ inv,
                 float* __restrict__ out)
{
    int t = blockIdx.x;
    int c = threadIdx.x * 4;
    int i0 = inv[t * TOP_K + 0];
    int i1 = inv[t * TOP_K + 1];
    float4 a = *(const float4*)&y[(size_t)i0 * D_OUT + c];
    float4 b = *(const float4*)&y[(size_t)i1 * D_OUT + c];
    float4 o = make_float4(a.x + b.x, a.y + b.y, a.z + b.z, a.w + b.w);
    *(float4*)&out[(size_t)t * D_OUT + c] = o;
}

// ---------------- host entry ----------------
extern "C" void kernel_launch(void* const* d_in, const int* in_sizes, int n_in,
                              void* d_out, int out_size, void* d_ws, size_t ws_size,
                              hipStream_t stream) {
    const float* x     = (const float*)d_in[0];
    const float* w     = (const float*)d_in[1];
    // d_in[2] = k (device scalar; TOP_K hardcoded to match reference)
    const int*   ssi   = (const int*)d_in[4];
    const int*   eoff  = (const int*)d_in[5];
    const float* gates = (const float*)d_in[6];
    float* out = (float*)d_out;

    // workspace layout (bytes)
    char* p = (char*)d_ws;
    float* y    = (float*)p;                 p += (size_t)NG * D_OUT * 4;         // 64 MB
    int*   inv  = (int*)p;                   p += (size_t)NG * 4;                  // 64 KB
    u16*   xhi  = (u16*)p;                   p += (size_t)(N_TOK + 1) * D_IN * 2;  // 16 MB (+zero row)
    u16*   xlo  = (u16*)p;                   p += (size_t)(N_TOK + 1) * D_IN * 2;
    u16*   whiT = (u16*)p;                   p += (size_t)N_EXP * D_IN * D_OUT * 2;// 16 MB
    u16*   wloT = (u16*)p;

    moe_convert_x<<<(N_TOK + 1) * (D_IN / 4) / 256, 256, 0, stream>>>(x, xhi, xlo);
    moe_convert_w<<<dim3(D_IN / 64, D_OUT / 64, N_EXP), 256, 0, stream>>>(w, whiT, wloT);
    moe_build_inv<<<NG / 256, 256, 0, stream>>>(ssi, inv, NG);

    dim3 grid(MAX_ROW_TILES, D_OUT / BN);
    moe_grouped_gemm<<<grid, 256, 0, stream>>>(xhi, xlo, whiT, wloT, ssi, eoff, gates, y);

    moe_combine<<<N_TOK, 256, 0, stream>>>(y, inv, out);
}
